// NasRNN_352187318638
// MI455X (gfx1250) — matrix operand order repaired and checked by reading, not compiled, not run on hardware
//
#include <hip/hip_runtime.h>
#include <hip/hip_bf16.h>
#include <stdint.h>

typedef __attribute__((ext_vector_type(16))) __bf16        v16bf;
typedef __attribute__((ext_vector_type(8)))  float         v8f;
typedef __attribute__((ext_vector_type(4)))  unsigned int  v4u;
typedef __attribute__((ext_vector_type(2)))  unsigned int  v2u;

union Frag { v16bf v; v4u q[2]; };

#define SEQ    128
#define BATCH  512
#define INDIM  512
#define HID    512
#define PLANE  (BATCH * HID)   // 262144 elements per [B,H] plane

// LDS row stride (elements) for staged tiles: 40 bf16 = 80 B = 20 banks/row,
// gcd(20,64)=4 -> 16 consecutive rows land in 16 distinct bank groups.
#define LDSW 40

// ---------------------------------------------------------------------------
// Setup kernel 1: fp32 -> bf16 convert (vectorized x4) for the inputs tensor.
// ---------------------------------------------------------------------------
__global__ __launch_bounds__(256) void cvt_bf16_x4(const float* __restrict__ in,
                                                   __bf16* __restrict__ out) {
  int i = blockIdx.x * 256 + threadIdx.x;          // one float4 per thread
  const float4 vv = ((const float4*)in)[i];
  union { __bf16 b[4]; v2u u; } t;
  t.b[0] = (__bf16)vv.x; t.b[1] = (__bf16)vv.y;
  t.b[2] = (__bf16)vv.z; t.b[3] = (__bf16)vv.w;
  *(v2u*)(out + 4ull * i) = t.u;
}

// ---------------------------------------------------------------------------
// Setup kernel 2: per-k 512x512 transpose + fp32->bf16.
// in : [8][512][512]  (k, i, h)     out: [8][512][512]  (k, h, i)
// ---------------------------------------------------------------------------
__global__ __launch_bounds__(256) void transpose_cvt_bf16(const float* __restrict__ in,
                                                          __bf16* __restrict__ out) {
  __shared__ __bf16 tile[32][33];
  const int k  = blockIdx.z;
  const int i0 = blockIdx.y * 32;
  const int h0 = blockIdx.x * 32;
  const int tx = threadIdx.x & 31;
  const int ty = threadIdx.x >> 5;                 // 0..7
  const float*  src = in  + (size_t)k * 512 * 512;
  __bf16*       dst = out + (size_t)k * 512 * 512;
  #pragma unroll
  for (int r = 0; r < 32; r += 8)
    tile[ty + r][tx] = (__bf16)src[(size_t)(i0 + ty + r) * 512 + (h0 + tx)];
  __syncthreads();
  #pragma unroll
  for (int r = 0; r < 32; r += 8)
    dst[(size_t)(h0 + ty + r) * 512 + (i0 + tx)] = tile[tx][ty + r];
}

// ---------------------------------------------------------------------------
// Setup kernel 3: init states: c = 1.0f (fp32), m = 1.0 (bf16).
// ---------------------------------------------------------------------------
__global__ __launch_bounds__(256) void init_state(float* __restrict__ c,
                                                  __bf16* __restrict__ mbf) {
  int i = blockIdx.x * 256 + threadIdx.x;
  c[i]   = 1.0f;
  mbf[i] = (__bf16)1.0f;
}

// ---------------------------------------------------------------------------
// Per-step fused GEMM:  s[k] = x_t @ Wih[k] + m @ Whh[k]   (K = 512 + 512)
//
// Block = 256 threads = 8 waves (4 M x 2 N), block tile 128(M) x 64(N),
// wave tile 32x32 (2x2 v_wmma_f32_16x16x32_bf16 accumulators).
// Each K=32 chunk of A (128x32) and B (64x32) is staged once into LDS
// (double-buffered, bank-conflict-free 40-element row stride) and shared by
// all 8 waves: 12 KB of global traffic per 0.52 MFLOP of WMMA work.
// Grid: x = N/64 (8), y = M/128 (4), z = k (8)  -> 256 blocks/step.
// ---------------------------------------------------------------------------
__global__ __launch_bounds__(256) void step_gemm(
    const __bf16* __restrict__ xt,     // [512][512] bf16 (b, i) for this t
    const __bf16* __restrict__ mbf,    // [512][512] bf16 (b, h)
    const __bf16* __restrict__ wihT,   // [8][512][512] bf16 (k, h, i)
    const __bf16* __restrict__ whhT,   // [8][512][512] bf16 (k, h, h')
    float* __restrict__ s)             // [8][512][512] fp32
{
  __shared__ __bf16 As[2][128 * LDSW];
  __shared__ __bf16 Bs[2][ 64 * LDSW];

  const int tid  = threadIdx.x;
  const int k    = blockIdx.z;
  const int lane = tid & 31;
  const int wave = tid >> 5;               // 0..7
  const int hi   = lane >> 4;              // 0: lanes 0-15, 1: lanes 16-31
  const int l15  = lane & 15;
  const int m0   = blockIdx.y * 128;       // block batch-row base
  const int n0   = blockIdx.x * 64;        // block hid-col  base
  const int moff = (wave >> 1) * 32;       // wave row offset in block tile
  const int noff = (wave &  1) * 32;       // wave col offset in block tile

  const __bf16* wA = wihT + (size_t)k * 512 * 512;
  const __bf16* wB = whhT + (size_t)k * 512 * 512;

  // Staging assignments (per thread): A: 2 threads/row x 16 elems,
  // B: 4 threads/row x 8 elems.
  const int arow = tid >> 1, ahalf = tid & 1;   // A: row 0..127, 16-elem half
  const int brow = tid >> 2, bq    = tid & 3;   // B: row 0..63,  8-elem quarter

  v4u pa0, pa1, pb;                        // prefetch registers

  auto loadg = [&](int ch) {
    const int half = ch >> 4;              // 0: x/Wih, 1: m/Whh
    const int kk   = (ch & 15) << 5;       // K offset within the 512 half
    const __bf16* Ag = (half ? mbf : xt) +
                       (size_t)(m0 + arow) * 512 + kk + ahalf * 16;
    pa0 = *(const v4u*)(Ag);
    pa1 = *(const v4u*)(Ag + 8);
    const __bf16* Bg = (half ? wB : wA) +
                       (size_t)(n0 + brow) * 512 + kk + bq * 8;
    pb  = *(const v4u*)(Bg);
  };
  auto storel = [&](int buf) {
    *(v4u*)(&As[buf][arow * LDSW + ahalf * 16])     = pa0;
    *(v4u*)(&As[buf][arow * LDSW + ahalf * 16 + 8]) = pa1;
    *(v4u*)(&Bs[buf][brow * LDSW + bq * 8])         = pb;
  };

  v8f acc[2][2] = {};

  loadg(0);
  storel(0);
  __syncthreads();

  for (int ch = 0; ch < 32; ++ch) {
    const int buf = ch & 1;
    if (ch < 31) loadg(ch + 1);            // prefetch next chunk (global)

    // Fragment reads from LDS (all 16 B aligned -> ds_load_b128).
    Frag a[2], b[2];
    #pragma unroll
    for (int i = 0; i < 2; ++i) {
      const __bf16* pa = &As[buf][(moff + i * 16 + l15) * LDSW + hi * 8];
      a[i].q[0] = *(const v4u*)(pa);
      a[i].q[1] = *(const v4u*)(pa + 16);
    }
    #pragma unroll
    for (int j = 0; j < 2; ++j) {
      const __bf16* pbl = &Bs[buf][(noff + j * 16 + l15) * LDSW + hi * 8];
      b[j].q[0] = *(const v4u*)(pbl);
      b[j].q[1] = *(const v4u*)(pbl + 16);
    }

    acc[0][0] = __builtin_amdgcn_wmma_f32_16x16x32_bf16(
        false, a[0].v, false, b[0].v, (short)0, acc[0][0], false, false);
    acc[0][1] = __builtin_amdgcn_wmma_f32_16x16x32_bf16(
        false, a[0].v, false, b[1].v, (short)0, acc[0][1], false, false);
    acc[1][0] = __builtin_amdgcn_wmma_f32_16x16x32_bf16(
        false, a[1].v, false, b[0].v, (short)0, acc[1][0], false, false);
    acc[1][1] = __builtin_amdgcn_wmma_f32_16x16x32_bf16(
        false, a[1].v, false, b[1].v, (short)0, acc[1][1], false, false);

    if (ch < 31) {
      storel(buf ^ 1);                     // safe: barrier after last read of buf^1
      __syncthreads();
    }
  }

  // C/D 16x16 f32 layout: VGPR v holds row (v + hi*8), col = l15.
  float* so = s + (size_t)k * PLANE;
  #pragma unroll
  for (int i = 0; i < 2; ++i) {
    #pragma unroll
    for (int j = 0; j < 2; ++j) {
      const int row = m0 + moff + i * 16 + hi * 8;
      const int col = n0 + noff + j * 16 + l15;
      #pragma unroll
      for (int v = 0; v < 8; ++v)
        so[(size_t)(row + v) * 512 + col] = acc[i][j][v];
    }
  }
}

// ---------------------------------------------------------------------------
// Per-step elementwise NAS cell.  Reads s[0..7] and c; writes new c (in place),
// new m as fp32 straight into d_out, and new m as bf16 for the next GEMM.
// ---------------------------------------------------------------------------
__device__ __forceinline__ float sigmoidf(float x) { return 1.0f / (1.0f + __expf(-x)); }

__global__ __launch_bounds__(256) void step_cell(const float* __restrict__ s,
                                                 float* __restrict__ c,
                                                 float* __restrict__ m_out,
                                                 __bf16* __restrict__ mbf) {
  const int idx = blockIdx.x * 256 + threadIdx.x;
  const float s0 = s[0 * PLANE + idx];
  const float s1 = s[1 * PLANE + idx];
  const float s2 = s[2 * PLANE + idx];
  const float s3 = s[3 * PLANE + idx];
  const float s4 = s[4 * PLANE + idx];
  const float s5 = s[5 * PLANE + idx];
  const float s6 = s[6 * PLANE + idx];
  const float s7 = s[7 * PLANE + idx];

  const float l1_0 = sigmoidf(s0);
  const float l1_1 = fmaxf(s1, 0.0f);
  const float l1_2 = sigmoidf(s2);
  const float l1_3 = fmaxf(s3, 0.0f);
  const float l1_4 = tanhf(s4);
  const float l1_5 = sigmoidf(s5);
  const float l1_6 = tanhf(s6);
  const float l1_7 = sigmoidf(s7);

  const float l2_0 = tanhf(l1_0 * l1_1);
  const float l2_1 = tanhf(l1_2 + l1_3);
  const float l2_2 = tanhf(l1_4 * l1_5);
  const float l2_3 = sigmoidf(l1_6 + l1_7);

  const float l2_0v2 = tanhf(l2_0 + c[idx]);
  const float new_c  = l2_0v2 * l2_1;
  const float l3_1   = tanhf(l2_2 + l2_3);
  const float new_m  = tanhf(new_c * l3_1);

  c[idx]     = new_c;
  m_out[idx] = new_m;          // final step leaves the answer in d_out
  mbf[idx]   = (__bf16)new_m;  // bf16 state for next step's GEMM
}

// ---------------------------------------------------------------------------
// Host-side launcher (graph-capture safe: stream-only, no allocs/syncs).
// Workspace: xbf 64MB | wihT 4MB | whhT 4MB | c 1MB | mbf 0.5MB | s 8MB.
// ---------------------------------------------------------------------------
extern "C" void kernel_launch(void* const* d_in, const int* in_sizes, int n_in,
                              void* d_out, int out_size, void* d_ws, size_t ws_size,
                              hipStream_t stream) {
  const float* inputs = (const float*)d_in[0];   // [128][512][512]
  const float* w_ih   = (const float*)d_in[1];   // [8][512][512]
  const float* w_hh   = (const float*)d_in[2];   // [8][512][512]
  float* out = (float*)d_out;                    // [512][512]

  char* ws = (char*)d_ws;
  size_t off = 0;
  __bf16* xbf  = (__bf16*)(ws + off); off += (size_t)SEQ * PLANE * 2;
  __bf16* wihT = (__bf16*)(ws + off); off += (size_t)8 * 512 * 512 * 2;
  __bf16* whhT = (__bf16*)(ws + off); off += (size_t)8 * 512 * 512 * 2;
  float*  c    = (float*) (ws + off); off += (size_t)PLANE * 4;
  __bf16* mbf  = (__bf16*)(ws + off); off += (size_t)PLANE * 2;
  float*  s    = (float*) (ws + off); off += (size_t)8 * PLANE * 4;
  (void)off; (void)ws_size; (void)in_sizes; (void)n_in; (void)out_size;

  // One-time (per call) prep: bf16 inputs, transposed bf16 weights, init state.
  {
    const int n4 = SEQ * PLANE / 4;                       // float4 count
    cvt_bf16_x4<<<n4 / 256, 256, 0, stream>>>(inputs, xbf);
    transpose_cvt_bf16<<<dim3(16, 16, 8), 256, 0, stream>>>(w_ih, wihT);
    transpose_cvt_bf16<<<dim3(16, 16, 8), 256, 0, stream>>>(w_hh, whhT);
    init_state<<<PLANE / 256, 256, 0, stream>>>(c, mbf);
  }

  // Sequential scan: stream ordering carries the recurrence.
  for (int t = 0; t < SEQ; ++t) {
    const __bf16* xt = xbf + (size_t)t * PLANE;
    step_gemm<<<dim3(8, 4, 8), 256, 0, stream>>>(xt, mbf, wihT, whhT, s);
    step_cell<<<PLANE / 256, 256, 0, stream>>>(s, c, out, mbf);
  }
}
